// GNCAModel_10995116278155
// MI455X (gfx1250) — matrix operand order repaired and checked by reading, not compiled, not run on hardware
//
#include <hip/hip_runtime.h>
#include <hip/hip_bf16.h>
#include <math.h>

// ---------------------------------------------------------------------------
// GNCA model for MI455X (gfx1250, wave32, WMMA).
//
// Pipeline (all node-level GEMMs in bf16 WMMA w/ f32 accumulate):
//   1. xb   = pad16->32(bf16(x))
//   2. t    = relu(xb @ w1 + b1)          [WMMA, K=32]
//   3. h    = t @ w2 + b2                 [WMMA, K=256]  (bf16 out)
//   4. M    = relu(h @ conv_w + conv_b)   [WMMA, K=256]  (bf16 out, 25.6MB -> L2 resident)
//   5. agg[dst] += M[src]                 [gather bf16, f32 atomics, all L2 resident]
//   6. hcat = [h | bf16(agg)]             (N x 512)
//   7. p    = relu(hcat @ post_w1 + b1)   [WMMA, K=512]
//   8. out  = tanh(p @ post_w2 + b2)      [WMMA, K=256, Nout=16, f32 out]
// ---------------------------------------------------------------------------

typedef __attribute__((ext_vector_type(16))) __bf16 v16bf;
typedef __attribute__((ext_vector_type(8)))  __bf16 bf16x8;
typedef __attribute__((ext_vector_type(8)))  float  v8f;

#define HID 256

// ---------------------------------------------------------------------------
// Prep kernels
// ---------------------------------------------------------------------------

// src: K x Nout f32 row-major  ->  dst: Nout x Kpad bf16 row-major (transposed,
// zero padded in K). Transposed so WMMA B-fragments are contiguous per lane.
__global__ void k_transpose_to_bf16(const float* __restrict__ src,
                                    __bf16* __restrict__ dst,
                                    int K, int Nout, int Kpad) {
  int idx = blockIdx.x * blockDim.x + threadIdx.x;
  int total = Nout * Kpad;
  if (idx >= total) return;
  int n = idx / Kpad;
  int k = idx % Kpad;
  float v = (k < K) ? src[(size_t)k * Nout + n] : 0.0f;
  dst[idx] = (__bf16)v;
}

// x: N x 16 f32  ->  xb: N x 32 bf16 (K zero-padded 16 -> 32 for the x32 WMMA)
__global__ void k_pad_convert_x(const float* __restrict__ x,
                                __bf16* __restrict__ xb, int n_total) {
  int idx = blockIdx.x * blockDim.x + threadIdx.x;
  if (idx >= n_total) return;
  int i = idx >> 5;
  int k = idx & 31;
  xb[idx] = (k < 16) ? (__bf16)x[i * 16 + k] : (__bf16)0.0f;
}

__global__ void k_zero_f32(float* __restrict__ p, size_t n) {
  size_t idx = (size_t)blockIdx.x * blockDim.x + threadIdx.x;
  size_t stride = (size_t)gridDim.x * blockDim.x;
  for (size_t i = idx; i < n; i += stride) p[i] = 0.0f;
}

// hcat: N x 512 bf16 = [ h_bf16 (N x 256) | bf16(agg_f32) (N x 256) ]
__global__ void k_build_hcat(const __bf16* __restrict__ h,
                             const float* __restrict__ agg,
                             __bf16* __restrict__ hcat, int n_total) {
  int idx = blockIdx.x * blockDim.x + threadIdx.x;
  if (idx >= n_total) return;
  int i = idx >> 9;        // / 512
  int k = idx & 511;
  hcat[idx] = (k < HID) ? h[(size_t)i * HID + k]
                        : (__bf16)agg[(size_t)i * HID + (k - HID)];
}

// ---------------------------------------------------------------------------
// bf16 WMMA GEMM:  C[16 x 16*NBLK per wave] = act(A @ Wt^T + bias)
//   A : Mrows x K   bf16 row-major (lda == K)
//   Wt: Nout  x K   bf16 row-major (pre-transposed weight)
//   ACT: 0 = none, 1 = relu, 2 = tanh
// One wave (32 threads) per block; grid = (Mrows/16, Nout/(16*NBLK)).
// A/B fragments both use contiguous k = k0 + 16*half + s; f32 C/D layout per CDNA5 ISA 7.12.2.
// ---------------------------------------------------------------------------
template <int NBLK, int ACT, bool OUT_BF16>
__global__ __launch_bounds__(32)
void k_gemm_bf16_wmma(const __bf16* __restrict__ A,
                      const __bf16* __restrict__ Wt,
                      const float* __restrict__ bias,
                      void* __restrict__ Cout,
                      int K, int Nout, int Mrows) {
  const int lane = threadIdx.x;
  const int half = lane >> 4;   // which 16-lane half of the wave
  const int r    = lane & 15;
  const int row0 = blockIdx.x * 16;
  const int col0 = blockIdx.y * (16 * NBLK);

  v8f acc[NBLK] = {};

  // A-fragment source row (clamped for ragged M; stores are guarded below)
  int arow_idx = row0 + r;
  if (arow_idx >= Mrows) arow_idx = Mrows - 1;
  const __bf16* arow = A + (size_t)arow_idx * K;

  for (int k0 = 0; k0 < K; k0 += 32) {
    // A 16x32 bf16 fragment: elems 0..7 <- K = k0+half*16+0..7,
    //                        elems 8..15 <- K = k0+half*16+8..15
    bf16x8 alo = *(const bf16x8*)(arow + k0 + half * 16);
    bf16x8 ahi = *(const bf16x8*)(arow + k0 + half * 16 + 8);
    v16bf afrag;
#pragma unroll
    for (int i = 0; i < 8; ++i) {
      afrag[i]     = alo[i];
      afrag[8 + i] = ahi[i];
    }

#pragma unroll
    for (int j = 0; j < NBLK; ++j) {
      // B 32x16 bf16 fragment: lane holds column col0+j*16+r,
      // K = k0 + half*16 + 0..15 (contiguous in transposed W)
      const __bf16* brow =
          Wt + (size_t)(col0 + j * 16 + r) * K + k0 + half * 16;
      v16bf bfrag = *(const v16bf*)brow;
      acc[j] = __builtin_amdgcn_wmma_f32_16x16x32_bf16(
          false, afrag, false, bfrag, (short)0, acc[j], false, false);
    }
  }

  // Epilogue: C/D element (M = v + 8*half, N = lane%16) per VGPR v
#pragma unroll
  for (int j = 0; j < NBLK; ++j) {
    int col = col0 + j * 16 + r;
    float b = bias[col];
#pragma unroll
    for (int v = 0; v < 8; ++v) {
      int m = row0 + half * 8 + v;
      if (m < Mrows) {
        float val = acc[j][v] + b;
        if (ACT == 1) val = fmaxf(val, 0.0f);
        if (ACT == 2) val = tanhf(val);
        if (OUT_BF16)
          ((__bf16*)Cout)[(size_t)m * Nout + col] = (__bf16)val;
        else
          ((float*)Cout)[(size_t)m * Nout + col] = val;
      }
    }
  }
}

// ---------------------------------------------------------------------------
// Edge scatter: one wave per edge. agg[dst] += fp32(M_bf16[src]).
// M (25.6MB bf16) and agg (51.2MB f32) both fit in the 192MB L2, so this phase
// is L2-bandwidth bound, not HBM bound.
// ---------------------------------------------------------------------------
__global__ __launch_bounds__(256)
void k_edge_scatter(const int* __restrict__ src, const int* __restrict__ dst,
                    const __bf16* __restrict__ M, float* __restrict__ agg,
                    int nEdges) {
  int wave  = (int)((blockIdx.x * blockDim.x + threadIdx.x) >> 5);
  int lane  = threadIdx.x & 31;
  int nWave = (int)((gridDim.x * blockDim.x) >> 5);
  for (int e = wave; e < nEdges; e += nWave) {
    int s = src[e];
    int d = dst[e];
    bf16x8 m = *(const bf16x8*)(M + (size_t)s * HID + lane * 8);
    float* ap = agg + (size_t)d * HID + lane * 8;
#pragma unroll
    for (int i = 0; i < 8; ++i) {
      float v = (float)m[i];
      (void)__hip_atomic_fetch_add(ap + i, v, __ATOMIC_RELAXED,
                                   __HIP_MEMORY_SCOPE_AGENT);
    }
  }
}

// ---------------------------------------------------------------------------
// Host launcher
// ---------------------------------------------------------------------------
static inline size_t align_up(size_t v, size_t a) { return (v + a - 1) & ~(a - 1); }

extern "C" void kernel_launch(void* const* d_in, const int* in_sizes, int n_in,
                              void* d_out, int out_size, void* d_ws, size_t ws_size,
                              hipStream_t stream) {
  const float* x       = (const float*)d_in[0];
  const int*   eidx    = (const int*)d_in[1];
  const float* pre_w1  = (const float*)d_in[2];
  const float* pre_b1  = (const float*)d_in[3];
  const float* pre_w2  = (const float*)d_in[4];
  const float* pre_b2  = (const float*)d_in[5];
  const float* conv_w  = (const float*)d_in[6];
  const float* conv_b  = (const float*)d_in[7];
  const float* post_w1 = (const float*)d_in[8];
  const float* post_b1 = (const float*)d_in[9];
  const float* post_w2 = (const float*)d_in[10];
  const float* post_b2 = (const float*)d_in[11];
  float* out = (float*)d_out;

  const int N  = in_sizes[0] / 16;      // 50000
  const int E  = in_sizes[1] / 2;       // 800000
  const int F_OUT = 16;
  const int rowTiles = (N + 15) / 16;   // 3125

  const int* e_src = eidx;
  const int* e_dst = eidx + E;

  // ---- workspace carve-up ----
  char* ws = (char*)d_ws;
  size_t off = 0;
  auto carve = [&](size_t bytes) {
    size_t o = off;
    off = align_up(off + bytes, 256);
    return (void*)(ws + o);
  };
  __bf16* w1t  = (__bf16*)carve((size_t)HID * 32 * 2);        // 256 x 32
  __bf16* w2t  = (__bf16*)carve((size_t)HID * HID * 2);       // 256 x 256
  __bf16* cwt  = (__bf16*)carve((size_t)HID * HID * 2);       // 256 x 256
  __bf16* p1t  = (__bf16*)carve((size_t)HID * 2 * HID * 2);   // 256 x 512
  __bf16* p2t  = (__bf16*)carve((size_t)F_OUT * HID * 2);     // 16  x 256
  __bf16* xb   = (__bf16*)carve((size_t)N * 32 * 2);          // N x 32
  __bf16* tbuf = (__bf16*)carve((size_t)N * HID * 2);         // N x 256 (t, later p)
  __bf16* hbuf = (__bf16*)carve((size_t)N * HID * 2);         // N x 256
  __bf16* Mbuf = (__bf16*)carve((size_t)N * HID * 2);         // N x 256
  float*  agg  = (float*) carve((size_t)N * HID * 4);         // N x 256 f32
  __bf16* hcat = (__bf16*)carve((size_t)N * 2 * HID * 2);     // N x 512
  (void)ws_size;

  const int T = 256;
  // 1) weight prep: transpose + bf16 convert (zero-pad pre_w1 K 16->32)
  k_transpose_to_bf16<<<(HID * 32 + T - 1) / T, T, 0, stream>>>(pre_w1, w1t, 16, HID, 32);
  k_transpose_to_bf16<<<(HID * HID + T - 1) / T, T, 0, stream>>>(pre_w2, w2t, HID, HID, HID);
  k_transpose_to_bf16<<<(HID * HID + T - 1) / T, T, 0, stream>>>(conv_w, cwt, HID, HID, HID);
  k_transpose_to_bf16<<<(HID * 2 * HID + T - 1) / T, T, 0, stream>>>(post_w1, p1t, 2 * HID, HID, 2 * HID);
  k_transpose_to_bf16<<<(F_OUT * HID + T - 1) / T, T, 0, stream>>>(post_w2, p2t, HID, F_OUT, HID);

  // 2) activations prep
  k_pad_convert_x<<<((N * 32) + T - 1) / T, T, 0, stream>>>(x, xb, N * 32);
  k_zero_f32<<<4096, T, 0, stream>>>(agg, (size_t)N * HID);

  // 3) pre-MLP
  dim3 g2(rowTiles, 2);   // 8 tiles * 16 cols = 128 cols per wave, 2 col groups
  k_gemm_bf16_wmma<8, 1, true><<<g2, 32, 0, stream>>>(xb,   w1t, pre_b1, tbuf, 32,  HID, N);
  k_gemm_bf16_wmma<8, 0, true><<<g2, 32, 0, stream>>>(tbuf, w2t, pre_b2, hbuf, HID, HID, N);

  // 4) per-node message transform (gather moved AFTER the GEMM)
  k_gemm_bf16_wmma<8, 1, true><<<g2, 32, 0, stream>>>(hbuf, cwt, conv_b, Mbuf, HID, HID, N);

  // 5) edge gather / scatter-add (L2 resident)
  k_edge_scatter<<<(E + 7) / 8, T, 0, stream>>>(e_src, e_dst, Mbuf, agg, E);

  // 6) concat [h | agg] as bf16
  k_build_hcat<<<((N * 2 * HID) + T - 1) / T, T, 0, stream>>>(hbuf, agg, hcat, N * 2 * HID);

  // 7) post-MLP (+ tanh, f32 output)
  k_gemm_bf16_wmma<8, 1, true><<<g2, 32, 0, stream>>>(hcat, p1t, post_b1, tbuf, 2 * HID, HID, N);
  dim3 g1(rowTiles, 1);
  k_gemm_bf16_wmma<1, 2, false><<<g1, 32, 0, stream>>>(tbuf, p2t, post_b2, out, HID, F_OUT, N);
  (void)out_size; (void)n_in;
}
